// CrossAttentionBlock_76587856822509
// MI455X (gfx1250) — matrix operand order repaired and checked by reading, not compiled, not run on hardware
//
#include <hip/hip_runtime.h>

// Problem constants (match reference)
#define LQ    2048
#define LKV   2048
#define CDIM  1024
#define CTX   768
#define NH    16
#define DKH   64
#define BATCH 2
#define HD    (NH * DKH)   // 1024
#define KVW   (2 * HD)     // 2048
#define SCALEF 0.125f      // 64^-0.5

typedef __attribute__((ext_vector_type(16))) __bf16 v16bf;
typedef __attribute__((ext_vector_type(8)))  float  v8f;

union FragU { v16bf v; uint4 u[2]; };
union U4BF  { uint4 u; __bf16 b[8]; };

__device__ inline __bf16 f2bf(float f) {
  union { float f; unsigned u; } x; x.f = f;
  unsigned r = (x.u + 0x7FFFu + ((x.u >> 16) & 1u)) >> 16;
  union { unsigned short s; __bf16 b; } y; y.s = (unsigned short)r;
  return y.b;
}
__device__ inline unsigned pack2(float a, float b) {
  union { __bf16 b; unsigned short s; } lo, hi;
  lo.b = f2bf(a); hi.b = f2bf(b);
  return (unsigned)lo.s | ((unsigned)hi.s << 16);
}

// A fragment (16x32 bf16, M x K). base -> element [0][0]; stride in elements.
// ISA layout: lane row = lane&15; lanes 0-15 hold K 0-7 & 16-23, lanes 16-31 K 8-15 & 24-31.
__device__ inline v16bf load_a_frag(const __bf16* base, int stride, int lane) {
  int row = lane & 15, kh = lane >> 4;
  FragU f;
  f.u[0] = *(const uint4*)(base + row * stride + kh * 8);
  f.u[1] = *(const uint4*)(base + row * stride + 16 + kh * 8);
  return f.v;
}
// B fragment (32x16 bf16, K x N) from an [N][K] transposed LDS tile.
// ISA layout: lane col = lane&15; lanes 0-15 hold K 0-7 & 16-23, lanes 16-31 K 8-15 & 24-31.
__device__ inline v16bf load_b_frag(const __bf16* baseT, int stride, int lane) {
  int n = lane & 15, kh = lane >> 4;
  FragU f;
  f.u[0] = *(const uint4*)(baseT + n * stride + kh * 8);
  f.u[1] = *(const uint4*)(baseT + n * stride + 16 + kh * 8);
  return f.v;
}
__device__ inline v8f wmma_bf16(v16bf a, v16bf b, v8f c) {
  return __builtin_amdgcn_wmma_f32_16x16x32_bf16(false, a, false, b, (short)0, c, false, false);
}

// Async global->LDS copy of 16 bytes per lane (GVS mode), tracked by ASYNCcnt.
__device__ inline void async_copy_b128(unsigned lds_off, int glb_off_bytes,
                                       const void* sbase) {
  asm volatile("global_load_async_to_lds_b128 %0, %1, %2"
               :: "v"(lds_off), "v"(glb_off_bytes), "s"(sbase)
               : "memory");
}
__device__ inline void wait_asynccnt0() {
  asm volatile("s_wait_asynccnt 0x0" ::: "memory");
}

__device__ inline float rowmax16(float v) {
#pragma unroll
  for (int m = 1; m < 16; m <<= 1) v = fmaxf(v, __shfl_xor(v, m, 32));
  return v;
}
__device__ inline float rowsum16(float v) {
#pragma unroll
  for (int m = 1; m < 16; m <<= 1) v += __shfl_xor(v, m, 32);
  return v;
}

// ---------------------------------------------------------------------------
// Tiled GEMM: C[M,N] = A[M,K] @ B[K,N]  (bf16 WMMA, f32 accumulate)
// Block tile 128x128, BK=64, 256 threads = 8 waves (2x4), wave tile 64x32.
// A_BF16: A operand already bf16 (else f32, converted on load).
// OUT_F32_BIAS: write f32 with fused bias (else write bf16).
// ---------------------------------------------------------------------------
template <bool A_BF16, bool OUT_F32_BIAS>
__global__ __launch_bounds__(256) void gemm_wmma_kernel(
    const void* __restrict__ Aptr, const float* __restrict__ Bw,
    const float* __restrict__ bias, void* __restrict__ Cout,
    int K, int lda, int ldb, int ldc) {
  __shared__ __bf16 As[128][72];    // [m][k], stride 72 -> 16B aligned, conflict-free
  __shared__ __bf16 BsT[128][72];   // [n][k]

  const int tid = threadIdx.x;
  const int lane = tid & 31, wid = tid >> 5;
  const int wm = (wid & 1) * 64, wn = (wid >> 1) * 32;
  const int bm = blockIdx.x * 128, bn = blockIdx.y * 128;

  v8f acc[4][2];
#pragma unroll
  for (int mt = 0; mt < 4; ++mt)
#pragma unroll
    for (int nt = 0; nt < 2; ++nt)
#pragma unroll
      for (int r = 0; r < 8; ++r) acc[mt][nt][r] = 0.0f;

  const int ar0 = tid >> 4, ac4 = (tid & 15) * 4;   // A: 16 rows/pass, 4 cols each
  const int bk0 = tid >> 5, bn4 = (tid & 31) * 4;   // B: 8 k/pass, 4 n each

  for (int k0 = 0; k0 < K; k0 += 64) {
    __syncthreads();
    // ---- stage A (128 x 64) ----
    if (A_BF16) {
      const __bf16* A = (const __bf16*)Aptr;
#pragma unroll
      for (int p = 0; p < 8; ++p) {
        int r = ar0 + p * 16;
        uint2 d = *(const uint2*)(A + (size_t)(bm + r) * lda + k0 + ac4);
        *(uint2*)&As[r][ac4] = d;
      }
      if (k0 + 64 < K)
        __builtin_prefetch(A + (size_t)(bm + ar0) * lda + k0 + 64 + ac4, 0, 1);
    } else {
      const float* A = (const float*)Aptr;
#pragma unroll
      for (int p = 0; p < 8; ++p) {
        int r = ar0 + p * 16;
        float4 d = *(const float4*)(A + (size_t)(bm + r) * lda + k0 + ac4);
        *(uint2*)&As[r][ac4] = make_uint2(pack2(d.x, d.y), pack2(d.z, d.w));
      }
      if (k0 + 64 < K)
        __builtin_prefetch(A + (size_t)(bm + ar0) * lda + k0 + 64 + ac4, 0, 1);
    }
    // ---- stage B transposed (64 x 128 -> [n][k]) ----
#pragma unroll
    for (int p = 0; p < 8; ++p) {
      int kk = bk0 + p * 8;
      float4 d = *(const float4*)(Bw + (size_t)(k0 + kk) * ldb + bn + bn4);
      BsT[bn4 + 0][kk] = f2bf(d.x);
      BsT[bn4 + 1][kk] = f2bf(d.y);
      BsT[bn4 + 2][kk] = f2bf(d.z);
      BsT[bn4 + 3][kk] = f2bf(d.w);
    }
    if (k0 + 64 < K)
      __builtin_prefetch(Bw + (size_t)(k0 + 64 + bk0) * ldb + bn + bn4, 0, 1);
    __syncthreads();

    // ---- compute: two 32-wide K chunks, 4x2 WMMA tiles per wave ----
#pragma unroll
    for (int dc = 0; dc < 2; ++dc) {
      v16bf afr[4], bfr[2];
#pragma unroll
      for (int mt = 0; mt < 4; ++mt)
        afr[mt] = load_a_frag(&As[wm + mt * 16][dc * 32], 72, lane);
#pragma unroll
      for (int nt = 0; nt < 2; ++nt)
        bfr[nt] = load_b_frag(&BsT[wn + nt * 16][dc * 32], 72, lane);
#pragma unroll
      for (int mt = 0; mt < 4; ++mt)
#pragma unroll
        for (int nt = 0; nt < 2; ++nt)
          acc[mt][nt] = wmma_bf16(afr[mt], bfr[nt], acc[mt][nt]);
    }
  }

  // ---- epilogue (C/D layout: row = r + 8*half, col = lane&15) ----
  const int hf = lane >> 4, nl = lane & 15;
#pragma unroll
  for (int mt = 0; mt < 4; ++mt)
#pragma unroll
    for (int nt = 0; nt < 2; ++nt)
#pragma unroll
      for (int r = 0; r < 8; ++r) {
        int row = bm + wm + mt * 16 + r + hf * 8;
        int col = bn + wn + nt * 16 + nl;
        float v = acc[mt][nt][r];
        if (OUT_F32_BIAS)
          ((float*)Cout)[(size_t)row * ldc + col] = v + bias[col];
        else
          ((__bf16*)Cout)[(size_t)row * ldc + col] = f2bf(v);
      }
}

// ---------------------------------------------------------------------------
// Flash attention: one workgroup = 64 query rows of one (b,h); 4 waves,
// each wave owns 16 query rows. KV streamed in 64-key tiles through LDS.
// Q/K tiles staged with GLOBAL_LOAD_ASYNC_TO_LDS_B128 (ASYNCcnt); V goes
// through VGPRs because it is transposed into [d][j] on the way in.
// S = Q K^T and O += P V both via v_wmma_f32_16x16x32_bf16.
// ---------------------------------------------------------------------------
__global__ __launch_bounds__(128) void flash_attn_kernel(
    const __bf16* __restrict__ qb,   // [B*LQ, HD]
    const __bf16* __restrict__ kvb,  // [B*LKV, 2*HD] : [K|V] interleaved blocks
    __bf16* __restrict__ resb) {     // [B*LQ, HD]
  __shared__ __bf16 Qs[64][72];        // [q][d]
  __shared__ __bf16 Ks[64][72];        // [j][d]  (== B^T layout for QK^T)
  __shared__ __bf16 VsT[64][72];       // [d][j]  (== B^T layout for PV)
  __shared__ __bf16 Ps[4][16][72];     // per-wave P staging (C-layout -> A-layout)

  const int tid = threadIdx.x;
  const int lane = tid & 31, wid = tid >> 5;
  const int hf = lane >> 4, nl = lane & 15;
  const int bh = blockIdx.y;
  const int b = bh >> 4, h = bh & 15;
  const int q0 = blockIdx.x * 64;

  const int lr = tid >> 3;           // 0..15  (16 rows per pass)
  const int c8 = (tid & 7) * 8;      // 8-element column chunk

  // ---- async-load Q tile (64 x 64): 16B per lane per issue ----
#pragma unroll
  for (int p = 0; p < 4; ++p) {
    int r = lr + p * 16;
    int goff = (int)(((size_t)(b * LQ + q0 + r) * HD + h * DKH + c8) *
                     sizeof(__bf16));
    async_copy_b128((unsigned)(size_t)&Qs[r][c8], goff, qb);
  }

  float mrow[8], lrow[8];
  v8f oacc[4];
#pragma unroll
  for (int r = 0; r < 8; ++r) { mrow[r] = -1e30f; lrow[r] = 0.0f; }
#pragma unroll
  for (int dt = 0; dt < 4; ++dt)
#pragma unroll
    for (int r = 0; r < 8; ++r) oacc[dt][r] = 0.0f;

  for (int j0 = 0; j0 < LKV; j0 += 64) {
    __syncthreads();   // previous iteration's consumers are done with Ks/VsT
    // ---- stage K tile [j][d] (async) and V tile transposed [d][j] ----
#pragma unroll
    for (int p = 0; p < 4; ++p) {
      int j = lr + p * 16;
      size_t rowbase = (size_t)(b * LKV + j0 + j) * KVW + h * DKH;
      async_copy_b128((unsigned)(size_t)&Ks[j][c8],
                      (int)((rowbase + c8) * sizeof(__bf16)), kvb);
      U4BF vd;
      vd.u = *(const uint4*)(kvb + rowbase + HD + c8);
#pragma unroll
      for (int i = 0; i < 8; ++i) VsT[c8 + i][j] = vd.b[i];
    }
    wait_asynccnt0();  // this wave's Q/K copies have landed in LDS
    __syncthreads();   // so have everyone else's

    // ---- S = Q K^T (wave: 16 q x 64 j), scaled ----
    v16bf qa[2];
#pragma unroll
    for (int dc = 0; dc < 2; ++dc)
      qa[dc] = load_a_frag(&Qs[wid * 16][dc * 32], 72, lane);
    v8f s[4];
#pragma unroll
    for (int jt = 0; jt < 4; ++jt) {
#pragma unroll
      for (int r = 0; r < 8; ++r) s[jt][r] = 0.0f;
#pragma unroll
      for (int dc = 0; dc < 2; ++dc)
        s[jt] = wmma_bf16(qa[dc], load_b_frag(&Ks[jt * 16][dc * 32], 72, lane), s[jt]);
#pragma unroll
      for (int r = 0; r < 8; ++r) s[jt][r] *= SCALEF;
    }

    // ---- online softmax update ----
#pragma unroll
    for (int r = 0; r < 8; ++r) {
      float t = fmaxf(fmaxf(s[0][r], s[1][r]), fmaxf(s[2][r], s[3][r]));
      float mnew = fmaxf(mrow[r], rowmax16(t));
      float corr = __expf(mrow[r] - mnew);
      mrow[r] = mnew;
      lrow[r] *= corr;
#pragma unroll
      for (int dt = 0; dt < 4; ++dt) oacc[dt][r] *= corr;
      float psum = 0.0f;
#pragma unroll
      for (int jt = 0; jt < 4; ++jt) {
        s[jt][r] = __expf(s[jt][r] - mnew);
        psum += s[jt][r];
      }
      lrow[r] += rowsum16(psum);
    }

    // ---- P: C-layout -> A-layout via wave-private LDS tile ----
#pragma unroll
    for (int jt = 0; jt < 4; ++jt)
#pragma unroll
      for (int r = 0; r < 8; ++r)
        Ps[wid][r + hf * 8][jt * 16 + nl] = f2bf(s[jt][r]);

    // ---- O += P V ----
    v16bf pa[2];
#pragma unroll
    for (int jc = 0; jc < 2; ++jc)
      pa[jc] = load_a_frag(&Ps[wid][0][jc * 32], 72, lane);
#pragma unroll
    for (int dt = 0; dt < 4; ++dt)
#pragma unroll
      for (int jc = 0; jc < 2; ++jc)
        oacc[dt] = wmma_bf16(pa[jc], load_b_frag(&VsT[dt * 16][jc * 32], 72, lane), oacc[dt]);
  }

  // ---- normalize and write out (bf16, [B*LQ, HD] layout for out-proj GEMM) ----
#pragma unroll
  for (int r = 0; r < 8; ++r) {
    float inv = 1.0f / lrow[r];
    int row = q0 + wid * 16 + r + hf * 8;
    __bf16* orow = resb + (size_t)(b * LQ + row) * HD + h * DKH;
#pragma unroll
    for (int dt = 0; dt < 4; ++dt)
      orow[dt * 16 + nl] = f2bf(oacc[dt][r] * inv);
  }
}

// ---------------------------------------------------------------------------
// Host launcher: Q proj -> KV proj -> flash attention -> out proj (+bias)
// Workspace: qb (8MB bf16) | kvb (16MB bf16) | resb (8MB bf16)
// ---------------------------------------------------------------------------
extern "C" void kernel_launch(void* const* d_in, const int* in_sizes, int n_in,
                              void* d_out, int out_size, void* d_ws, size_t ws_size,
                              hipStream_t stream) {
  const float* x     = (const float*)d_in[0];
  const float* y     = (const float*)d_in[1];
  const float* Wq    = (const float*)d_in[2];
  const float* Wkv   = (const float*)d_in[3];
  const float* Wproj = (const float*)d_in[4];
  const float* bproj = (const float*)d_in[5];

  __bf16* qb   = (__bf16*)d_ws;
  __bf16* kvb  = qb + (size_t)BATCH * LQ * HD;
  __bf16* resb = kvb + (size_t)BATCH * LKV * KVW;

  // Q = x @ Wq          : M=4096, N=1024, K=1024
  gemm_wmma_kernel<false, false><<<dim3((BATCH * LQ) / 128, HD / 128), 256, 0, stream>>>(
      x, Wq, nullptr, qb, CDIM, CDIM, HD, HD);
  // KV = y @ Wkv        : M=4096, N=2048, K=768
  gemm_wmma_kernel<false, false><<<dim3((BATCH * LKV) / 128, KVW / 128), 256, 0, stream>>>(
      y, Wkv, nullptr, kvb, CTX, CTX, KVW, KVW);
  // attention           : 32 (b,h) x 32 query tiles
  flash_attn_kernel<<<dim3(LQ / 64, BATCH * NH), 128, 0, stream>>>(qb, kvb, resb);
  // out = res @ Wproj + b : M=4096, N=1024, K=1024
  gemm_wmma_kernel<true, true><<<dim3((BATCH * LQ) / 128, CDIM / 128), 256, 0, stream>>>(
      resb, Wproj, bproj, (float*)d_out, HD, HD, CDIM, CDIM);
}
